// single_head_attention_27187142984351
// MI455X (gfx1250) — compile-verified, matrix operand order checked by reading, not run on hardware
//
#include <hip/hip_runtime.h>
#include <hip/hip_bf16.h>
#include <math.h>

typedef __bf16 bf16;
typedef __attribute__((ext_vector_type(16))) __bf16 v16bf;
typedef __attribute__((ext_vector_type(8)))  float  v8f;
typedef unsigned int uint32;

constexpr int DIN   = 2048;   // input dim
constexpr int DOUT  = 512;    // head dim
constexpr int WLEN  = 2048;   // sequence length
constexpr int BATCH = 8;

// LDS tile rows hold 32 K-elements, padded to 40 bf16 (80 B) per row:
// keeps 16-byte alignment for b128 fragment loads and staggers banks.
constexpr int LDR  = 40;   // bf16 elements per LDS row
constexpr int LDRD = 20;   // dwords per LDS row

// ---------------------------------------------------------------------------
// WMMA fragment load from LDS, 16-bit A/B layout per CDNA5 ISA 7.12.2:
//   lanes 0-15  : row = lane,    K = 0..7  (v0..v3), K = 16..23 (v4..v7)
//   lanes 16-31 : row = lane-16, K = 8..15 (v0..v3), K = 24..31 (v4..v7)
// Source is row-major [row][K] with row stride LDR. Two ds_load_b128 per lane.
// ---------------------------------------------------------------------------
__device__ __forceinline__ v16bf load_tile16(const bf16* base) {
  const int lane = threadIdx.x & 31;
  const int m    = lane & 15;
  const int kb   = (lane & 16) >> 1;          // 0 or 8
  const bf16* r  = base + m * LDR;
  union { v16bf v; uint4 q[2]; } u;
  u.q[0] = *(const uint4*)(r + kb);           // K = kb .. kb+7
  u.q[1] = *(const uint4*)(r + 16 + kb);      // K = 16+kb .. 16+kb+7
  return u.v;
}

__device__ __forceinline__ uint32 pack2bf(float x, float y) {
  union { bf16 h[2]; uint32 u; } p;
  p.h[0] = (bf16)x; p.h[1] = (bf16)y;
  return p.u;
}

__device__ __forceinline__ v8f zero8() {
  v8f z;
  for (int i = 0; i < 8; ++i) z[i] = 0.0f;
  return z;
}

// ---------------------------------------------------------------------------
// Projection GEMM:  Out[o,t] = sum_i Mw[o,i] * X[b,i,t]   (bf16 out)
// 128(o) x 128(t) block, k-chunks of 32 over i. Double-buffered LDS,
// global loads for chunk k+1 issued before the WMMA pass of chunk k.
// transposeOut=1 -> store OutT[t*DOUT + o]  (for Qd / Kd)
// transposeOut=0 -> store Out [o*WLEN + t]  (for Vd)
// ---------------------------------------------------------------------------
__global__ __launch_bounds__(256) void proj_gemm_kernel(
    const float* __restrict__ Mw, const float* __restrict__ X,
    bf16* __restrict__ Out, int transposeOut)
{
  __shared__ __align__(16) bf16 Abuf[2][128 * LDR];   // [o_local][i_local]
  __shared__ __align__(16) bf16 Bbuf[2][128 * LDR];   // [t_local][i_local]
  const int tid = threadIdx.x;
  const int wv  = tid >> 5;
  const int o0  = blockIdx.x * 128;
  const int t0  = blockIdx.y * 128;
  const float* Xb = X + (size_t)blockIdx.z * DIN * WLEN;
  bf16* Ob = Out + (size_t)blockIdx.z * (size_t)DOUT * WLEN;

  const int ra = tid >> 4;    // A-stage row base (0..15)
  const int ca = tid & 15;    // A-stage dword column
  const int tb = tid & 127;   // B-stage t column
  const int ib = tid >> 7;    // B-stage i-pair base (0..1)

  float2 aL[8];
  float  bL0[8], bL1[8];

  auto loadStage = [&](int kc) {
    const int i0 = kc * 32;
#pragma unroll
    for (int jj = 0; jj < 8; ++jj) {
      const int r = ra + 16 * jj;
      aL[jj] = *(const float2*)(Mw + (size_t)(o0 + r) * DIN + i0 + ca * 2);
      const int ip = ib + 2 * jj;
      const float* s = Xb + (size_t)(i0 + 2 * ip) * WLEN + t0 + tb;
      bL0[jj] = s[0];
      bL1[jj] = s[WLEN];
    }
  };
  auto storeStage = [&](int buf) {
    uint32* Ad = (uint32*)Abuf[buf];
    uint32* Bd = (uint32*)Bbuf[buf];
#pragma unroll
    for (int jj = 0; jj < 8; ++jj) {
      const int r = ra + 16 * jj;
      Ad[r * LDRD + ca] = pack2bf(aL[jj].x, aL[jj].y);
      const int ip = ib + 2 * jj;
      Bd[tb * LDRD + ip] = pack2bf(bL0[jj], bL1[jj]);
    }
  };

  v8f acc[8];
#pragma unroll
  for (int j = 0; j < 8; ++j) acc[j] = zero8();

  constexpr int NK = DIN / 32;
  loadStage(0);
  storeStage(0);
  __syncthreads();
  for (int kc = 0; kc < NK; ++kc) {
    const int cur = kc & 1;
    if (kc + 1 < NK) {
      loadStage(kc + 1);   // global loads in flight during WMMA pass
      __builtin_prefetch(Xb + (size_t)(kc * 32 + 64) * WLEN + t0 + tb, 0, 3);
    }
    v16bf a = load_tile16(&Abuf[cur][(wv * 16) * LDR]);
    v16bf bm[8];
#pragma unroll
    for (int j = 0; j < 8; ++j) bm[j] = load_tile16(&Bbuf[cur][(j * 16) * LDR]);
#pragma unroll
    for (int j = 0; j < 8; ++j)
      acc[j] = __builtin_amdgcn_wmma_f32_16x16x32_bf16(
          false, a, false, bm[j], (short)0, acc[j], false, false);
    if (kc + 1 < NK) storeStage(cur ^ 1);
    __syncthreads();
  }

  // C layout: VGPR r -> (m = r + 8*(lane>=16), n = lane&15)
  const int lane = tid & 31;
  const int n    = lane & 15;
  const int mh   = (lane & 16) ? 8 : 0;
#pragma unroll
  for (int j = 0; j < 8; ++j) {
#pragma unroll
    for (int r = 0; r < 8; ++r) {
      const int og = o0 + wv * 16 + r + mh;
      const int tg = t0 + j * 16 + n;
      const bf16 v = (bf16)acc[j][r];
      if (transposeOut) Ob[(size_t)tg * DOUT + og] = v;
      else              Ob[(size_t)og * WLEN + tg] = v;
    }
  }
}

// ---------------------------------------------------------------------------
// Score GEMM: S[s,t] = scale * sum_o KdT[s,o] * QdT[t,o]   (store only t<=s)
// Blocks fully above the diagonal are skipped. blockIdx.z = batch.
// ---------------------------------------------------------------------------
__global__ __launch_bounds__(256) void corr_gemm_kernel(
    const bf16* __restrict__ KdT, const bf16* __restrict__ QdT,
    float* __restrict__ S, size_t projStride, size_t sStride)
{
  if (blockIdx.y > blockIdx.x) return;   // whole block masked (t > s)
  __shared__ __align__(16) bf16 Abuf[2][128 * LDR];   // [s_local][o_local]
  __shared__ __align__(16) bf16 Bbuf[2][128 * LDR];   // [t_local][o_local]
  KdT += (size_t)blockIdx.z * projStride;
  QdT += (size_t)blockIdx.z * projStride;
  S   += (size_t)blockIdx.z * sStride;
  const int tid = threadIdx.x;
  const int wv  = tid >> 5;
  const int s0  = blockIdx.x * 128;
  const int t0  = blockIdx.y * 128;
  const int ra  = tid >> 4;
  const int ca  = tid & 15;

  uint32 aL[8], bL[8];
  auto loadStage = [&](int kc) {
    const int o0 = kc * 32;
#pragma unroll
    for (int jj = 0; jj < 8; ++jj) {
      const int r = ra + 16 * jj;
      aL[jj] = *(const uint32*)(KdT + (size_t)(s0 + r) * DOUT + o0 + ca * 2);
      bL[jj] = *(const uint32*)(QdT + (size_t)(t0 + r) * DOUT + o0 + ca * 2);
    }
  };
  auto storeStage = [&](int buf) {
    uint32* Ad = (uint32*)Abuf[buf];
    uint32* Bd = (uint32*)Bbuf[buf];
#pragma unroll
    for (int jj = 0; jj < 8; ++jj) {
      const int r = ra + 16 * jj;
      Ad[r * LDRD + ca] = aL[jj];
      Bd[r * LDRD + ca] = bL[jj];
    }
  };

  v8f acc[8];
#pragma unroll
  for (int j = 0; j < 8; ++j) acc[j] = zero8();

  constexpr int NK = DOUT / 32;
  loadStage(0);
  storeStage(0);
  __syncthreads();
  for (int kc = 0; kc < NK; ++kc) {
    const int cur = kc & 1;
    if (kc + 1 < NK) loadStage(kc + 1);
    v16bf a = load_tile16(&Abuf[cur][(wv * 16) * LDR]);
    v16bf bm[8];
#pragma unroll
    for (int j = 0; j < 8; ++j) bm[j] = load_tile16(&Bbuf[cur][(j * 16) * LDR]);
#pragma unroll
    for (int j = 0; j < 8; ++j)
      acc[j] = __builtin_amdgcn_wmma_f32_16x16x32_bf16(
          false, a, false, bm[j], (short)0, acc[j], false, false);
    if (kc + 1 < NK) storeStage(cur ^ 1);
    __syncthreads();
  }

  const float scale = 0.044194173824159216f;  // 1/sqrt(512)
  const int lane = tid & 31;
  const int n    = lane & 15;
  const int mh   = (lane & 16) ? 8 : 0;
#pragma unroll
  for (int j = 0; j < 8; ++j) {
#pragma unroll
    for (int r = 0; r < 8; ++r) {
      const int sg = s0 + wv * 16 + r + mh;
      const int tg = t0 + j * 16 + n;
      if (tg <= sg) S[(size_t)sg * WLEN + tg] = acc[j][r] * scale;
    }
  }
}

// ---------------------------------------------------------------------------
// Row softmax over t in [0, s]; writes P[s,t] bf16 (zeros for t > s).
// One 256-thread workgroup per row; blockIdx.y = batch.
// ---------------------------------------------------------------------------
__global__ __launch_bounds__(256) void col_softmax_kernel(
    const float* __restrict__ S, bf16* __restrict__ P,
    size_t sStride, size_t pStride)
{
  S += (size_t)blockIdx.y * sStride;
  P += (size_t)blockIdx.y * pStride;
  const int s   = blockIdx.x;
  const int tid = threadIdx.x;
  const int n   = s + 1;                 // valid length
  __shared__ float red[256];

  float m = -INFINITY;
  for (int t = tid; t < n; t += 256) m = fmaxf(m, S[(size_t)s * WLEN + t]);
  red[tid] = m; __syncthreads();
  for (int o = 128; o > 0; o >>= 1) {
    if (tid < o) red[tid] = fmaxf(red[tid], red[tid + o]);
    __syncthreads();
  }
  m = red[0]; __syncthreads();

  float sum = 0.0f;
  for (int t = tid; t < n; t += 256) sum += __expf(S[(size_t)s * WLEN + t] - m);
  red[tid] = sum; __syncthreads();
  for (int o = 128; o > 0; o >>= 1) {
    if (tid < o) red[tid] += red[tid + o];
    __syncthreads();
  }
  const float inv = 1.0f / red[0];

  for (int t = tid; t < WLEN; t += 256) {
    const float p = (t < n) ? __expf(S[(size_t)s * WLEN + t] - m) * inv : 0.0f;
    P[(size_t)s * WLEN + t] = (bf16)p;
  }
}

// ---------------------------------------------------------------------------
// Output GEMM: Out[o,s] = sum_t P[s,t] * Vd[o,t]
// k-loop truncated at the causal boundary (P is zero past it). z = batch.
// ---------------------------------------------------------------------------
__global__ __launch_bounds__(256) void out_gemm_kernel(
    const bf16* __restrict__ P, const bf16* __restrict__ Vd,
    float* __restrict__ Out, size_t pStride, size_t projStride,
    size_t outStride)
{
  __shared__ __align__(16) bf16 Abuf[2][128 * LDR];   // [s_local][t_local]
  __shared__ __align__(16) bf16 Bbuf[2][128 * LDR];   // [o_local][t_local]
  P   += (size_t)blockIdx.z * pStride;
  Vd  += (size_t)blockIdx.z * projStride;
  Out += (size_t)blockIdx.z * outStride;
  const int tid = threadIdx.x;
  const int wv  = tid >> 5;
  const int s0  = blockIdx.x * 128;
  const int o0  = blockIdx.y * 128;
  const int kcN = (s0 >> 5) + 4;         // t <= s0+127 only
  const int ra  = tid >> 4;
  const int ca  = tid & 15;

  uint32 aL[8], bL[8];
  auto loadStage = [&](int kc) {
    const int tk = kc * 32;
#pragma unroll
    for (int jj = 0; jj < 8; ++jj) {
      const int r = ra + 16 * jj;
      aL[jj] = *(const uint32*)(P  + (size_t)(s0 + r) * WLEN + tk + ca * 2);
      bL[jj] = *(const uint32*)(Vd + (size_t)(o0 + r) * WLEN + tk + ca * 2);
    }
  };
  auto storeStage = [&](int buf) {
    uint32* Ad = (uint32*)Abuf[buf];
    uint32* Bd = (uint32*)Bbuf[buf];
#pragma unroll
    for (int jj = 0; jj < 8; ++jj) {
      const int r = ra + 16 * jj;
      Ad[r * LDRD + ca] = aL[jj];
      Bd[r * LDRD + ca] = bL[jj];
    }
  };

  v8f acc[8];
#pragma unroll
  for (int j = 0; j < 8; ++j) acc[j] = zero8();

  loadStage(0);
  storeStage(0);
  __syncthreads();
  for (int kc = 0; kc < kcN; ++kc) {
    const int cur = kc & 1;
    if (kc + 1 < kcN) loadStage(kc + 1);
    v16bf a = load_tile16(&Abuf[cur][(wv * 16) * LDR]);
    v16bf bm[8];
#pragma unroll
    for (int j = 0; j < 8; ++j) bm[j] = load_tile16(&Bbuf[cur][(j * 16) * LDR]);
#pragma unroll
    for (int j = 0; j < 8; ++j)
      acc[j] = __builtin_amdgcn_wmma_f32_16x16x32_bf16(
          false, a, false, bm[j], (short)0, acc[j], false, false);
    if (kc + 1 < kcN) storeStage(cur ^ 1);
    __syncthreads();
  }

  const int lane = tid & 31;
  const int n    = lane & 15;
  const int mh   = (lane & 16) ? 8 : 0;
#pragma unroll
  for (int j = 0; j < 8; ++j) {
#pragma unroll
    for (int r = 0; r < 8; ++r) {
      const int sg = s0 + wv * 16 + r + mh;
      const int og = o0 + j * 16 + n;
      Out[(size_t)og * WLEN + sg] = acc[j][r];
    }
  }
}

// ---------------------------------------------------------------------------
// Host launch
// ---------------------------------------------------------------------------
extern "C" void kernel_launch(void* const* d_in, const int* in_sizes, int n_in,
                              void* d_out, int out_size, void* d_ws, size_t ws_size,
                              hipStream_t stream) {
  (void)in_sizes; (void)n_in; (void)out_size;
  const float* Q   = (const float*)d_in[0];
  const float* Kin = (const float*)d_in[1];
  const float* V   = (const float*)d_in[2];
  const float* Mq  = (const float*)d_in[3];
  const float* Mk  = (const float*)d_in[4];
  const float* Mv  = (const float*)d_in[5];
  float* out = (float*)d_out;

  // workspace layout:
  //   QdT, KdT : (B, W, DOUT) bf16   Vd : (B, DOUT, W) bf16
  //   S : (zb, W, W) fp32   P : (zb, W, W) bf16
  // zb = 8 (all batches in flight) if ws allows (~252 MB), else 1 (~76 MB).
  const size_t projElems = (size_t)BATCH * WLEN * DOUT;   // per tensor
  const size_t swElems   = (size_t)WLEN * WLEN;
  const size_t projBytes = 3 * projElems * sizeof(bf16);
  const int zb = (ws_size >= projBytes + (size_t)BATCH * swElems * 6) ? BATCH : 1;

  bf16*  QdT = (bf16*)d_ws;
  bf16*  KdT = QdT + projElems;
  bf16*  Vd  = KdT + projElems;
  float* S   = (float*)(Vd + projElems);
  bf16*  P   = (bf16*)(S + (size_t)zb * swElems);

  const size_t pStride   = (size_t)WLEN * DOUT;   // per-batch proj stride
  const size_t outStride = (size_t)DOUT * WLEN;

  const dim3 blk(256);
  const dim3 gproj(DOUT / 128, WLEN / 128, BATCH);
  proj_gemm_kernel<<<gproj, blk, 0, stream>>>(Mq, Q,   QdT, 1);
  proj_gemm_kernel<<<gproj, blk, 0, stream>>>(Mk, Kin, KdT, 1);
  proj_gemm_kernel<<<gproj, blk, 0, stream>>>(Mv, V,   Vd,  0);

  for (int b0 = 0; b0 < BATCH; b0 += zb) {
    const size_t po = (size_t)b0 * pStride;
    corr_gemm_kernel<<<dim3(WLEN / 128, WLEN / 128, zb), blk, 0, stream>>>(
        KdT + po, QdT + po, S, pStride, swElems);
    col_softmax_kernel<<<dim3(WLEN, zb), blk, 0, stream>>>(
        S, P, swElems, swElems);
    out_gemm_kernel<<<dim3(WLEN / 128, DOUT / 128, zb), blk, 0, stream>>>(
        P, Vd + po, out + (size_t)b0 * outStride, swElems, pStride, outStride);
  }
}
